// MoELayer_26439818674680
// MI455X (gfx1250) — matrix operand order repaired and checked by reading, not compiled, not run on hardware
//
#include <hip/hip_runtime.h>
#include <hip/hip_bf16.h>
#include <math.h>

// ---------------------------------------------------------------------------
// MoE forward, top-2 routing. Compute-bound (409 FLOP/byte) -> use f16 WMMA
// (v_wmma_f32_16x16x32_f16) with fp32 accumulate. Weights pre-converted to
// fp16 in workspace; 64-token tiles; double-buffered LDS weight panels.
// N=4096 tokens, C=1024, D=4096, E=8.
// ---------------------------------------------------------------------------

typedef __attribute__((ext_vector_type(16))) _Float16 v16h;
typedef __attribute__((ext_vector_type(8)))  _Float16 v8h;
typedef __attribute__((ext_vector_type(8)))  float    v8f;

#define N_TOK   4096
#define C_DIM   1024
#define D_DIM   4096
#define E_NUM   8

#define TM        64      // tokens per block
#define DC        128     // D chunk
#define KP        128     // K panel (both stages)
#define NTHREADS  512     // 16 waves
#define SXK       1032    // sX stride (halves): 516 dwords, %64==4 -> conflict-free
#define SWK       136     // weight-panel / sH stride (halves): 68 dwords, %64==4
#define SWBUF     (128 * SWK)   // 17408 halves per panel buffer
#define SMEM_HALVES (TM * SXK + 2 * SWBUF + TM * SWK)
#define SMEM_BYTES  (SMEM_HALVES * 2)

__device__ __forceinline__ v16h cat8(v8h lo, v8h hi) {
    return __builtin_shufflevector(lo, hi, 0,1,2,3,4,5,6,7,8,9,10,11,12,13,14,15);
}
__device__ __forceinline__ v8f wmma16(v16h a, v16h b, v8f c) {
    return __builtin_amdgcn_wmma_f32_16x16x32_f16(false, a, false, b,
                                                  (short)0, c, false, false);
}

// ---------------------------------------------------------------------------
__global__ __launch_bounds__(256) void zero_kernel(float* out, int total, int* counts) {
    if (blockIdx.x == 0 && threadIdx.x < E_NUM) counts[threadIdx.x] = 0;
    int stride = gridDim.x * blockDim.x;
    for (int i = blockIdx.x * blockDim.x + threadIdx.x; i < total; i += stride)
        out[i] = 0.0f;
}

// fp32 -> fp16 bulk convert (float4 -> 4 halves)
__global__ __launch_bounds__(256) void cvt_kernel(const float* __restrict__ src,
                                                  _Float16* __restrict__ dst, int n4) {
    typedef __attribute__((ext_vector_type(4))) _Float16 h4;
    int stride = gridDim.x * blockDim.x;
    for (int i = blockIdx.x * blockDim.x + threadIdx.x; i < n4; i += stride) {
        float4 v = ((const float4*)src)[i];
        h4 o = {(_Float16)v.x, (_Float16)v.y, (_Float16)v.z, (_Float16)v.w};
        ((h4*)dst)[i] = o;
    }
}

// ---------------------------------------------------------------------------
// Router: logits = x @ wr^T, top-2; renormalized softmax over 2 == sigmoid.
__global__ __launch_bounds__(256) void router_kernel(const float* __restrict__ x,
                                                     const float* __restrict__ wr,
                                                     int* __restrict__ counts,
                                                     int* __restrict__ toks,
                                                     float* __restrict__ wgts) {
    __shared__ float swr[E_NUM * C_DIM];
    int tid = threadIdx.x;
    for (int i = tid; i < E_NUM * C_DIM; i += 256) swr[i] = wr[i];
    __syncthreads();

    int n = blockIdx.x * 256 + tid;
    const float* xr = x + (size_t)n * C_DIM;
    float lg[E_NUM];
#pragma unroll
    for (int e = 0; e < E_NUM; ++e) lg[e] = 0.0f;
    for (int c = 0; c < C_DIM; c += 4) {
        float4 xv = *(const float4*)(xr + c);
#pragma unroll
        for (int e = 0; e < E_NUM; ++e) {
            const float* w = swr + e * C_DIM + c;
            lg[e] += xv.x * w[0] + xv.y * w[1] + xv.z * w[2] + xv.w * w[3];
        }
    }
    int i0 = 0;
#pragma unroll
    for (int e = 1; e < E_NUM; ++e) if (lg[e] > lg[i0]) i0 = e;
    int i1 = (i0 == 0) ? 1 : 0;
#pragma unroll
    for (int e = 0; e < E_NUM; ++e) if (e != i0 && lg[e] > lg[i1]) i1 = e;
    float w0 = 1.0f / (1.0f + expf(lg[i1] - lg[i0]));
    float w1 = 1.0f - w0;

    int s0 = atomicAdd(&counts[i0], 1);
    toks[i0 * N_TOK + s0] = n;  wgts[i0 * N_TOK + s0] = w0;
    int s1 = atomicAdd(&counts[i1], 1);
    toks[i1 * N_TOK + s1] = n;  wgts[i1 * N_TOK + s1] = w1;
}

// ---------------------------------------------------------------------------
// Fused expert FFN, f16 WMMA. Block = 64 tokens of one expert, 16 waves.
__global__ __launch_bounds__(NTHREADS) void ffn_kernel(
        const _Float16* __restrict__ xh,
        const _Float16* __restrict__ w1h, const float* __restrict__ b1,
        const _Float16* __restrict__ w2h, const float* __restrict__ b2,
        const int* __restrict__ counts, const int* __restrict__ toks,
        const float* __restrict__ wgts, float* __restrict__ out) {
    extern __shared__ _Float16 smem[];
    _Float16* sX = smem;                     // [64][SXK]
    _Float16* sW = smem + TM * SXK;          // 2 x [128][SWK] ping-pong panels
    _Float16* sH = sW + 2 * SWBUF;           // [64][SWK]
    __shared__ int   sTok[TM];
    __shared__ float sWt[TM];

    const int e    = blockIdx.y;
    const int cnt  = counts[e];
    const int base = blockIdx.x * TM;
    if (base >= cnt) return;
    const int valid = min(TM, cnt - base);

    const int tid  = threadIdx.x;
    const int wv   = tid >> 5;       // wave 0..15
    const int lane = tid & 31;
    const int half = lane >> 4;
    const int l16  = lane & 15;
    const int mw   = wv & 3;         // M row-tile (16 rows each)
    const int ng   = wv >> 2;        // N group (32 cols each within a 128 panel)

    if (tid < TM) {
        int idx = base + tid;
        if (idx < cnt) { sTok[tid] = toks[(size_t)e * N_TOK + idx];
                         sWt[tid]  = wgts[(size_t)e * N_TOK + idx]; }
        else           { sTok[tid] = 0; sWt[tid] = 0.0f; }
    }
    __syncthreads();

    // Stage X tile [64 x 1024] fp16 into LDS (16B chunks).
    for (int i = tid; i < TM * (C_DIM / 8); i += NTHREADS) {
        int row = i >> 7;              // C/8 == 128 chunks per row
        int c8  = (i & 127) << 3;
        *(v8h*)(sX + (size_t)row * SXK + c8) =
            *(const v8h*)(xh + (size_t)sTok[row] * C_DIM + c8);
    }

    const _Float16* w1e = w1h + (size_t)e * C_DIM * D_DIM;
    const _Float16* w2e = w2h + (size_t)e * D_DIM * C_DIM;

    // Transpose-stage a [128k x 128n] fp16 panel into dst[n*SWK + k].
    auto stagePanel = [&](const _Float16* __restrict__ src, size_t srcStride,
                          _Float16* __restrict__ dst) {
#pragma unroll 4
        for (int i = tid; i < (KP * 128) / 2; i += NTHREADS) {   // 8192 b32 units
            int k  = i >> 6;
            int n2 = (i & 63) << 1;
            unsigned u = *(const unsigned*)(src + (size_t)k * srcStride + n2);
            dst[(size_t)(n2    ) * SWK + k] =
                __builtin_bit_cast(_Float16, (unsigned short)(u & 0xffffu));
            dst[(size_t)(n2 + 1) * SWK + k] =
                __builtin_bit_cast(_Float16, (unsigned short)(u >> 16));
        }
    };

    v8f outAcc[8][2] = {};   // [N panel][tile] -> cols 128*p + ng*32 + 16*t + l16

    for (int dchunk = 0; dchunk < D_DIM / DC; ++dchunk) {
        const int dc0 = dchunk * DC;

        // ---------------- Stage 1: H[64x128] = X @ W1[:, dc0:+128] ----------
        v8f hAcc[2] = {};
        stagePanel(w1e + 0 * (size_t)KP * D_DIM + dc0, D_DIM, sW);   // panel 0
        for (int kp = 0; kp < C_DIM / KP; ++kp) {
            __syncthreads();
            if (kp + 1 < C_DIM / KP)
                stagePanel(w1e + (size_t)(kp + 1) * KP * D_DIM + dc0, D_DIM,
                           sW + ((kp + 1) & 1) * SWBUF);
            const _Float16* wb = sW + (kp & 1) * SWBUF;
#pragma unroll
            for (int ks = 0; ks < 4; ++ks) {
                const _Float16* ap = sX + (size_t)(16 * mw + l16) * SXK
                                        + kp * KP + ks * 32 + 8 * half;
                v16h A = cat8(*(const v8h*)ap, *(const v8h*)(ap + 16));
#pragma unroll
                for (int t = 0; t < 2; ++t) {
                    const _Float16* bp = wb + (size_t)(ng * 32 + 16 * t + l16) * SWK
                                            + ks * 32 + 8 * half;
                    v16h B = cat8(*(const v8h*)bp, *(const v8h*)(bp + 16));
                    hAcc[t] = wmma16(A, B, hAcc[t]);
                }
            }
        }
        // bias + exact GELU -> sH (fp16)
#pragma unroll
        for (int t = 0; t < 2; ++t) {
            float bias = b1[e * D_DIM + dc0 + ng * 32 + 16 * t + l16];
#pragma unroll
            for (int r = 0; r < 8; ++r) {
                float v = hAcc[t][r] + bias;
                v = 0.5f * v * (1.0f + erff(v * 0.70710678118654752f));
                sH[(size_t)(16 * mw + r + 8 * half) * SWK
                   + ng * 32 + 16 * t + l16] = (_Float16)v;
            }
        }
        __syncthreads();   // sH visible; all stage-1 panel reads complete

        // ---------------- Stage 2: Out += H @ W2[dc0:+128, :] ----------------
        stagePanel(w2e + (size_t)dc0 * C_DIM + 0 * 128, C_DIM, sW);  // panel 0
        for (int p = 0; p < C_DIM / 128; ++p) {
            __syncthreads();
            if (p + 1 < C_DIM / 128)
                stagePanel(w2e + (size_t)dc0 * C_DIM + (p + 1) * 128, C_DIM,
                           sW + ((p + 1) & 1) * SWBUF);
            const _Float16* wb = sW + (p & 1) * SWBUF;
#pragma unroll
            for (int ks = 0; ks < 4; ++ks) {
                const _Float16* ap = sH + (size_t)(16 * mw + l16) * SWK
                                        + ks * 32 + 8 * half;
                v16h A = cat8(*(const v8h*)ap, *(const v8h*)(ap + 16));
#pragma unroll
                for (int t = 0; t < 2; ++t) {
                    const _Float16* bp = wb + (size_t)(ng * 32 + 16 * t + l16) * SWK
                                            + ks * 32 + 8 * half;
                    v16h B = cat8(*(const v8h*)bp, *(const v8h*)(bp + 16));
                    outAcc[p][t] = wmma16(A, B, outAcc[p][t]);
                }
            }
        }
        // no trailing barrier needed: next chunk's first stage/barrier covers hazards
    }

    // ---------------- Epilogue: +b2, weight, atomic scatter ------------------
#pragma unroll
    for (int p = 0; p < 8; ++p) {
#pragma unroll
        for (int t = 0; t < 2; ++t) {
            int col = 128 * p + ng * 32 + 16 * t + l16;
            float bb = b2[e * C_DIM + col];
#pragma unroll
            for (int r = 0; r < 8; ++r) {
                int row = 16 * mw + r + 8 * half;
                if (row < valid) {
                    float v = sWt[row] * (outAcc[p][t][r] + bb);
                    atomicAdd(&out[(size_t)sTok[row] * C_DIM + col], v);
                }
            }
        }
    }
}

// ---------------------------------------------------------------------------
extern "C" void kernel_launch(void* const* d_in, const int* in_sizes, int n_in,
                              void* d_out, int out_size, void* d_ws, size_t ws_size,
                              hipStream_t stream) {
    (void)in_sizes; (void)n_in; (void)out_size; (void)ws_size;
    const float* x  = (const float*)d_in[0];
    const float* wr = (const float*)d_in[1];
    const float* w1 = (const float*)d_in[2];
    const float* b1 = (const float*)d_in[3];
    const float* w2 = (const float*)d_in[4];
    const float* b2 = (const float*)d_in[5];
    float* out = (float*)d_out;

    char* ws = (char*)d_ws;
    size_t off = 0;
    int*   counts = (int*)(ws + off);  off += 256;
    int*   toks   = (int*)(ws + off);  off += (size_t)E_NUM * N_TOK * 4;
    float* wgts   = (float*)(ws + off);off += (size_t)E_NUM * N_TOK * 4;
    off = (off + 255) & ~(size_t)255;
    _Float16* xh  = (_Float16*)(ws + off); off += (size_t)N_TOK * C_DIM * 2;
    _Float16* w1h = (_Float16*)(ws + off); off += (size_t)E_NUM * C_DIM * D_DIM * 2;
    _Float16* w2h = (_Float16*)(ws + off); // off += same

    zero_kernel<<<1024, 256, 0, stream>>>(out, N_TOK * C_DIM, counts);
    cvt_kernel<<<2048, 256, 0, stream>>>(x,  xh,  N_TOK * C_DIM / 4);
    cvt_kernel<<<2048, 256, 0, stream>>>(w1, w1h, E_NUM * C_DIM * D_DIM / 4);
    cvt_kernel<<<2048, 256, 0, stream>>>(w2, w2h, E_NUM * D_DIM * C_DIM / 4);
    router_kernel<<<N_TOK / 256, 256, 0, stream>>>(x, wr, counts, toks, wgts);

    dim3 grid(N_TOK / TM, E_NUM);
    ffn_kernel<<<grid, NTHREADS, SMEM_BYTES, stream>>>(
        xh, w1h, b1, w2h, b2, counts, toks, wgts, out);
}
